// DeepseekV2AttentionSimple_73160472920712
// MI455X (gfx1250) — matrix operand order repaired and checked by reading, not compiled, not run on hardware
//
#include <hip/hip_runtime.h>
#include <hip/hip_bf16.h>

// ---------------- CDNA5 WMMA plumbing (wave32, gfx1250) ----------------
typedef __attribute__((ext_vector_type(16))) __bf16 v16bf;
typedef __attribute__((ext_vector_type(8)))  float  v8f;

union FragB16 { v16bf v; unsigned u[8]; };

__device__ __forceinline__ v8f vzero() {
  v8f z;
#pragma unroll
  for (int i = 0; i < 8; ++i) z[i] = 0.0f;
  return z;
}

__device__ __forceinline__ unsigned short f2bf(float f) {
  unsigned x = __float_as_uint(f);
  x += 0x7FFFu + ((x >> 16) & 1u);   // round-to-nearest-even
  return (unsigned short)(x >> 16);
}
__device__ __forceinline__ unsigned pack2bf(float a, float b) {
  return (unsigned)f2bf(a) | ((unsigned)f2bf(b) << 16);
}

__device__ __forceinline__ v8f wmma_bf16(const FragB16& a, const FragB16& b, v8f c) {
  // (neg_a, A, neg_b, B, c_mod, C, reuse_a, reuse_b)
  return __builtin_amdgcn_wmma_f32_16x16x32_bf16(false, a.v, false, b.v, (short)0, c,
                                                 false, false);
}

// ---- CDNA5 async global->LDS copy (ASYNCcnt path, CDNA5 ISA)
// LDS destination offset is the low 32 bits of the generic shared pointer
// (ISA 10.2); global address is a 64-bit VGPR pair.
__device__ __forceinline__ void async_b128(void* lds_dst, const void* gsrc) {
  unsigned ldsoff = (unsigned)(unsigned long long)lds_dst;
  asm volatile("global_load_async_to_lds_b128 %0, %1, off"
               :: "v"(ldsoff), "v"(gsrc) : "memory");
}
__device__ __forceinline__ void wait_async0() {
  asm volatile("s_wait_asynccnt 0x0" ::: "memory");
}

// A-matrix 16x32 bf16 fragment from a row-major [rows][RS] bf16 (ushort) tile.
// Layout: lanes 0-15 -> M=lane, K=0..15 ; lanes 16-31 -> K=16..31.
template<int RS>
__device__ __forceinline__ FragB16 load_a(const unsigned short* base, int row0,
                                          int k0, int lr, int lh) {
  FragB16 f;
#pragma unroll
  for (int i = 0; i < 8; ++i) {
    int kk = 2 * i + lh * 16;
    f.u[i] = *(const unsigned*)(base + (row0 + lr) * RS + k0 + kk);
  }
  return f;
}

// B-matrix 32x16 bf16 fragment where B is stored as [n][k] rows (k contiguous),
// i.e. C = A * B^T style. Lane lr -> column N, K pairs contiguous.
template<int RS>
__device__ __forceinline__ FragB16 load_b_rowk(const unsigned short* base, int col0,
                                               int k0, int lr, int lh) {
  FragB16 f;
#pragma unroll
  for (int i = 0; i < 8; ++i) {
    int kk = 2 * i + lh * 16;
    f.u[i] = *(const unsigned*)(base + (col0 + lr) * RS + k0 + kk);
  }
  return f;
}

// B-matrix fragment where B is stored as [k][n] with row stride colstride
// (k pairs gathered with stride).
__device__ __forceinline__ FragB16 load_b_colk(const unsigned short* base, int col,
                                               int k0, int colstride, int lr, int lh) {
  FragB16 f;
#pragma unroll
  for (int i = 0; i < 8; ++i) {
    int kk = k0 + 2 * i + lh * 16;
    unsigned lo = base[kk * colstride + col];
    unsigned hi = base[(kk + 1) * colstride + col];
    f.u[i] = lo | (hi << 16);
  }
  return f;
}

// ---------------- Generic WMMA GEMM: C = A * op(B) ----------------
// A: f32 [M,K] (lda). B: f32, b_nt=1 -> B is [N,K] (ldb=K stride), C=A*B^T;
//                      b_nt=0 -> B is [K,N] (ldb=N stride), C=A*B.
// C: f32 or bf16 (out_bf16), ldc. Batched: g -> (g/batch_h, g%batch_h) strides.
#define GBM 128
#define GBN 64
#define GBK 32

__global__ __launch_bounds__(256) void gemm_wmma_kernel(
    const float* __restrict__ A, const float* __restrict__ Bw, void* __restrict__ Cv,
    int M, int N, int K, int lda, int ldb, int ldc, int b_nt, int out_bf16,
    int batch_h,
    long long sA0, long long sA1, long long sB0, long long sB1,
    long long sC0, long long sC1) {
  __shared__ unsigned short As[GBM * GBK];
  __shared__ unsigned short Bs[GBN * GBK];
  const int tid = threadIdx.x;
  const int wid = tid >> 5, lane = tid & 31;
  const int lr = lane & 15, lh = lane >> 4;

  const int g = blockIdx.z;
  const int g0 = g / batch_h, g1 = g % batch_h;
  A += g0 * sA0 + g1 * sA1;
  Bw += g0 * sB0 + g1 * sB1;
  const long long coff = g0 * sC0 + g1 * sC1;

  const int m0 = blockIdx.y * GBM;
  const int n0 = blockIdx.x * GBN;
  const int wm = (wid & 3) * 32;   // 4 waves cover 128 rows
  const int wn = (wid >> 2) * 32;  // 2 waves cover 64 cols

  v8f acc[2][2];
#pragma unroll
  for (int mt = 0; mt < 2; ++mt)
#pragma unroll
    for (int nt = 0; nt < 2; ++nt) acc[mt][nt] = vzero();

  for (int k0 = 0; k0 < K; k0 += GBK) {
    // Stage A tile (128x32 f32 -> bf16): 16 elements/thread, coalesced on k.
#pragma unroll
    for (int j = 0; j < 16; ++j) {
      int e = j * 256 + tid;
      int r = e >> 5, c = e & 31;
      As[e] = f2bf(A[(long long)(m0 + r) * lda + (k0 + c)]);
    }
    // Stage B tile (64x32) as [n][k].
#pragma unroll
    for (int j = 0; j < 8; ++j) {
      int e = j * 256 + tid;
      int r = e >> 5, c = e & 31;
      float v = b_nt ? Bw[(long long)(n0 + r) * ldb + (k0 + c)]
                     : Bw[(long long)(k0 + c) * ldb + (n0 + r)];
      Bs[e] = f2bf(v);
    }
    if (k0 + GBK < K) {  // hint next K-slab into cache (global_prefetch_b8)
      __builtin_prefetch(&A[(long long)(m0 + (tid & 127)) * lda + k0 + GBK], 0, 3);
      __builtin_prefetch(&Bw[(long long)(tid & 63) * ldb + k0 + GBK], 0, 3);
    }
    __syncthreads();

    FragB16 bfr[2];
#pragma unroll
    for (int nt = 0; nt < 2; ++nt) bfr[nt] = load_b_rowk<GBK>(Bs, wn + nt * 16, 0, lr, lh);
#pragma unroll
    for (int mt = 0; mt < 2; ++mt) {
      FragB16 a = load_a<GBK>(As, wm + mt * 16, 0, lr, lh);
#pragma unroll
      for (int nt = 0; nt < 2; ++nt) acc[mt][nt] = wmma_bf16(a, bfr[nt], acc[mt][nt]);
    }
    __syncthreads();
  }

#pragma unroll
  for (int mt = 0; mt < 2; ++mt)
#pragma unroll
    for (int nt = 0; nt < 2; ++nt) {
      int col = n0 + wn + nt * 16 + lr;
#pragma unroll
      for (int r = 0; r < 8; ++r) {
        int row = m0 + wm + mt * 16 + lh * 8 + r;
        long long off = coff + (long long)row * ldc + col;
        if (out_bf16) ((unsigned short*)Cv)[off] = f2bf(acc[mt][nt][r]);
        else          ((float*)Cv)[off] = acc[mt][nt][r];
      }
    }
}

// ---------------- RMSNorm(ckv) + RoPE(k_pe) -> kv_cat bf16 [B,S,576] -------
// inv_freq[i] = 10000^(-2i/64) -> freq = pos * exp(-ln(1e4)/32 * i)
#define ROPE_LN (-0.2878231366242558f)

__global__ __launch_bounds__(256) void rmsnorm_rope_kv_kernel(
    const float* __restrict__ ckv_kpe, const float* __restrict__ lnw,
    const int* __restrict__ pos, unsigned short* __restrict__ kv_cat, int S_) {
  const int wid = threadIdx.x >> 5, lane = threadIdx.x & 31;
  const int row = blockIdx.x * 8 + wid;  // b*S + s
  const int s = row % S_;
  const float* src = ckv_kpe + (long long)row * 576;
  unsigned short* dst = kv_cat + (long long)row * 576;

  float v[16];
  float ss = 0.0f;
#pragma unroll
  for (int j = 0; j < 16; ++j) { v[j] = src[lane + j * 32]; ss += v[j] * v[j]; }
#pragma unroll
  for (int off = 16; off > 0; off >>= 1) ss += __shfl_xor(ss, off, 32);
  float inv = rsqrtf(ss * (1.0f / 512.0f) + 1e-6f);
#pragma unroll
  for (int j = 0; j < 16; ++j)
    dst[lane + j * 32] = f2bf(v[j] * inv * lnw[lane + j * 32]);

  // RoPE on k_pe (interleaved pairs).
  float t = (float)pos[s];
  float fi = t * __expf(ROPE_LN * (float)lane);
  float c = __cosf(fi), sn = __sinf(fi);
  float v0 = src[512 + 2 * lane], v1 = src[512 + 2 * lane + 1];
  dst[512 + lane]      = f2bf(v0 * c - v1 * sn);
  dst[512 + 32 + lane] = f2bf(v1 * c + v0 * sn);
}

// ---------------- RoPE(q_pe) -> q_cat[...,512:576] bf16 ----------------
__global__ __launch_bounds__(512) void rope_q_kernel(
    const float* __restrict__ q_pe, const int* __restrict__ pos,
    unsigned short* __restrict__ q_cat, int S_) {
  const int h = threadIdx.x >> 5, lane = threadIdx.x & 31;
  const int row = blockIdx.x;  // b*S + s
  const int b = row / S_, s = row % S_;
  const float* src = q_pe + (long long)row * 1024 + h * 64;
  float t = (float)pos[s];
  float fi = t * __expf(ROPE_LN * (float)lane);
  float c = __cosf(fi), sn = __sinf(fi);
  float v0 = src[2 * lane], v1 = src[2 * lane + 1];
  unsigned short* dst = q_cat + (((long long)(b * 16 + h) * S_) + s) * 576 + 512;
  dst[lane]      = f2bf(v0 * c - v1 * sn);
  dst[32 + lane] = f2bf(v1 * c + v0 * sn);
}

// ---------------- Flash-style MLA attention + fused v_b epilogue ----------
#define MQ 32
#define TK 64
#define DD 576
#define DKV 512
#define ATT_SCALE 0.07216878364870323f  // (128+64)^-0.5

struct AttnSmem {
  unsigned short Qs[MQ * DD];   // 36 KB
  unsigned short Ks[TK * DD];   // 72 KB (reused as O bf16 [MQ][DKV] in epilogue)
  float          Srow[MQ * TK]; //  8 KB
  unsigned short Pt[MQ * TK];   //  4 KB
  float rowm[MQ], rowl[MQ], rowf[MQ];
};

__global__ __launch_bounds__(256) void attn_kernel(
    const unsigned short* __restrict__ q_cat,
    const unsigned short* __restrict__ kv_cat,
    const float* __restrict__ v_b, float* __restrict__ v_out, int S_) {
  __shared__ AttnSmem sm;
  const int tid = threadIdx.x, wid = tid >> 5, lane = tid & 31;
  const int lr = lane & 15, lh = lane >> 4;
  const int qb = blockIdx.x, h = blockIdx.y, b = blockIdx.z;
  const int qbase = qb * MQ;

  // Stage Q tile (32 x 576 bf16 = 36 KB, contiguous) via async global->LDS.
  {
    const char* src =
        (const char*)(q_cat + ((long long)(b * 16 + h) * S_ + qbase) * DD);
    char* dst = (char*)sm.Qs;
#pragma unroll
    for (int j = 0; j < 9; ++j) {   // 2304 x b128 over 256 threads
      int byte = (j * 256 + tid) * 16;
      async_b128(dst + byte, src + byte);
    }
  }
  if (tid < MQ) { sm.rowm[tid] = -1e30f; sm.rowl[tid] = 0.0f; }

  v8f o[2][4];
#pragma unroll
  for (int mt = 0; mt < 2; ++mt)
#pragma unroll
    for (int nt = 0; nt < 4; ++nt) o[mt][nt] = vzero();

  const int nT = (qbase + MQ + TK - 1) / TK;
  for (int ti = 0; ti < nT; ++ti) {
    const int t0 = ti * TK;
    __syncthreads();  // protect Ks against previous iteration's readers
    {
      // K tile (64 x 576 bf16 = 72 KB, contiguous rows) via async global->LDS.
      const char* src = (const char*)(kv_cat + ((long long)b * S_ + t0) * DD);
      char* dst = (char*)sm.Ks;
#pragma unroll
      for (int j = 0; j < 18; ++j) {  // 4608 x b128 over 256 threads
        int byte = (j * 256 + tid) * 16;
        async_b128(dst + byte, src + byte);
      }
    }
    wait_async0();      // ASYNCcnt == 0: our async writes (and Q's) landed in LDS
    __syncthreads();    // make all waves' tiles visible

    // S tile 32x64: each wave owns one 16x16 tile (mt = wid&1, nt = wid>>1).
    {
      const int mt = wid & 1, nt = wid >> 1;
      v8f sacc = vzero();
#pragma unroll
      for (int k0 = 0; k0 < DD; k0 += 32) {
        FragB16 a  = load_a<DD>(sm.Qs, mt * 16, k0, lr, lh);
        FragB16 bb = load_b_rowk<DD>(sm.Ks, nt * 16, k0, lr, lh);
        sacc = wmma_bf16(a, bb, sacc);
      }
      const int colg = t0 + nt * 16 + lr;
#pragma unroll
      for (int r = 0; r < 8; ++r) {
        int rowloc = mt * 16 + lh * 8 + r;
        float sv = sacc[r] * ATT_SCALE;
        if (colg > qbase + rowloc) sv = -1e30f;  // causal mask
        sm.Srow[rowloc * TK + nt * 16 + lr] = sv;
      }
    }
    __syncthreads();

    if (tid < MQ) {  // online-softmax bookkeeping, one thread per query row
      float mx = sm.rowm[tid];
      const float* sr = &sm.Srow[tid * TK];
#pragma unroll
      for (int jj = 0; jj < TK; ++jj) mx = fmaxf(mx, sr[jj]);
      float f = __expf(sm.rowm[tid] - mx);
      sm.rowf[tid] = f;
      sm.rowl[tid] *= f;
      sm.rowm[tid] = mx;
    }
    __syncthreads();

    // P = exp(S - m), bf16 into LDS; row sums via ds_add_f32.
#pragma unroll
    for (int j = 0; j < 8; ++j) {
      int e = j * 256 + tid;  // MQ*TK = 2048
      int rr = e >> 6;
      float p = __expf(sm.Srow[e] - sm.rowm[rr]);
      sm.Pt[e] = f2bf(p);
      atomicAdd(&sm.rowl[rr], p);
    }
    // Rescale running O by exp(m_old - m_new).
#pragma unroll
    for (int mt = 0; mt < 2; ++mt)
#pragma unroll
      for (int r = 0; r < 8; ++r) {
        float f = sm.rowf[mt * 16 + lh * 8 + r];
#pragma unroll
        for (int nt = 0; nt < 4; ++nt) o[mt][nt][r] *= f;
      }
    __syncthreads();

    // O += P @ V; wave owns columns [wid*64, wid*64+64) of the 512 latent dims.
#pragma unroll
    for (int kt = 0; kt < 2; ++kt)
#pragma unroll
      for (int mt = 0; mt < 2; ++mt) {
        FragB16 a = load_a<TK>(sm.Pt, mt * 16, kt * 32, lr, lh);
#pragma unroll
        for (int nt = 0; nt < 4; ++nt) {
          int ccol = wid * 64 + nt * 16 + lr;
          FragB16 bb = load_b_colk(sm.Ks, ccol, kt * 32, DD, lr, lh);
          o[mt][nt] = wmma_bf16(a, bb, o[mt][nt]);
        }
      }
  }
  __syncthreads();

  // Normalize and stage O as bf16 [MQ][DKV] (reuse Ks).
  unsigned short* Os = sm.Ks;
#pragma unroll
  for (int mt = 0; mt < 2; ++mt)
#pragma unroll
    for (int nt = 0; nt < 4; ++nt) {
      int ccol = wid * 64 + nt * 16 + lr;
#pragma unroll
      for (int r = 0; r < 8; ++r) {
        int rowloc = mt * 16 + lh * 8 + r;
        Os[rowloc * DKV + ccol] = f2bf(o[mt][nt][r] / sm.rowl[rowloc]);
      }
    }
  __syncthreads();

  // Epilogue: Vout[32x128] = O @ v_b[h]^T  (v_b[h] is [128][512] f32).
  const float* vb = v_b + (long long)h * 128 * DKV;
  const int emt = wid & 1;
  const int en0 = (wid >> 1) * 2;  // two of 8 column tiles per wave
  v8f vo[2];
  vo[0] = vzero(); vo[1] = vzero();
  for (int k0 = 0; k0 < DKV; k0 += 32) {
    FragB16 a = load_a<DKV>(Os, emt * 16, k0, lr, lh);
#pragma unroll
    for (int j = 0; j < 2; ++j) {
      int n = (en0 + j) * 16 + lr;  // dv index
      FragB16 bb;
#pragma unroll
      for (int i = 0; i < 8; ++i) {
        int kk = k0 + 2 * i + lh * 16;
        const float* p = vb + (long long)n * DKV + kk;
        bb.u[i] = pack2bf(p[0], p[1]);
      }
      vo[j] = wmma_bf16(a, bb, vo[j]);
    }
  }
#pragma unroll
  for (int j = 0; j < 2; ++j) {
    int cc = h * 128 + (en0 + j) * 16 + lr;
#pragma unroll
    for (int r = 0; r < 8; ++r) {
      int rowloc = emt * 16 + lh * 8 + r;
      long long row = (long long)b * S_ + qbase + rowloc;
      v_out[row * 2048 + cc] = vo[j][r];
    }
  }
}

// ---------------- Host-side orchestration ----------------
extern "C" void kernel_launch(void* const* d_in, const int* in_sizes, int n_in,
                              void* d_out, int out_size, void* d_ws, size_t ws_size,
                              hipStream_t stream) {
  (void)in_sizes; (void)n_in; (void)out_size; (void)ws_size;
  const int B = 2, S = 2048, HID = 2048, H = 16;
  const float* hidden = (const float*)d_in[0];
  const float* q_nope_w = (const float*)d_in[1];  // [2048, 2048]
  const float* q_pe_w = (const float*)d_in[2];    // [1024, 2048]
  const float* kv_a_w = (const float*)d_in[3];    // [576, 2048]
  const float* kv_ln_w = (const float*)d_in[4];   // [512]
  const float* k_b_w = (const float*)d_in[5];     // [16,128,512]
  const float* v_b_w = (const float*)d_in[6];     // [16,128,512]
  const float* o_w = (const float*)d_in[7];       // [2048, 2048]
  const int* pos = (const int*)d_in[8];           // [S]
  float* out = (float*)d_out;

  const long long BS = (long long)B * S;  // 4096
  char* ws = (char*)d_ws;
  float* q_nope = (float*)ws;                  ws += BS * 2048 * sizeof(float);
  float* q_pe   = (float*)ws;                  ws += BS * 1024 * sizeof(float);
  float* ckv    = (float*)ws;                  ws += BS * 576 * sizeof(float);
  unsigned short* q_cat  = (unsigned short*)ws; ws += (long long)B * H * S * 576 * 2;
  unsigned short* kv_cat = (unsigned short*)ws; ws += BS * 576 * 2;
  float* v_out = (float*)ws;                   ws += BS * 2048 * sizeof(float);

  dim3 blk(256);
  // 1) q_nope = hidden @ q_nope_w^T   [4096,2048,2048]
  gemm_wmma_kernel<<<dim3(2048 / GBN, 4096 / GBM, 1), blk, 0, stream>>>(
      hidden, q_nope_w, q_nope, 4096, 2048, 2048, 2048, 2048, 2048, 1, 0, 1,
      0, 0, 0, 0, 0, 0);
  // 2) q_pe = hidden @ q_pe_w^T       [4096,1024,2048]
  gemm_wmma_kernel<<<dim3(1024 / GBN, 4096 / GBM, 1), blk, 0, stream>>>(
      hidden, q_pe_w, q_pe, 4096, 1024, 2048, 2048, 2048, 1024, 1, 0, 1,
      0, 0, 0, 0, 0, 0);
  // 3) ckv_kpe = hidden @ kv_a_w^T    [4096,576,2048]
  gemm_wmma_kernel<<<dim3(576 / GBN, 4096 / GBM, 1), blk, 0, stream>>>(
      hidden, kv_a_w, ckv, 4096, 576, 2048, 2048, 2048, 576, 1, 0, 1,
      0, 0, 0, 0, 0, 0);
  // 4) RMSNorm + RoPE(k_pe) -> kv_cat bf16
  rmsnorm_rope_kv_kernel<<<dim3((unsigned)(BS / 8)), blk, 0, stream>>>(
      ckv, kv_ln_w, pos, kv_cat, S);
  // 5) RoPE(q_pe) -> q_cat tail
  rope_q_kernel<<<dim3((unsigned)BS), dim3(512), 0, stream>>>(q_pe, pos, q_cat, S);
  // 6) q_lat per head: q_nope[b,:,h*128:+128] @ k_b[h] -> q_cat[...,0:512] bf16
  gemm_wmma_kernel<<<dim3(512 / GBN, 2048 / GBM, B * H), blk, 0, stream>>>(
      q_nope, k_b_w, q_cat, 2048, 512, 128, 2048, 512, 576, /*b_nt=*/0,
      /*out_bf16=*/1, /*batch_h=*/H,
      /*sA0=*/(long long)S * 2048, /*sA1=*/128,
      /*sB0=*/0, /*sB1=*/(long long)128 * 512,
      /*sC0=*/(long long)H * S * 576, /*sC1=*/(long long)S * 576);
  // 7) flash attention + fused v_b epilogue -> v_out [B*S, H*128]
  attn_kernel<<<dim3(S / MQ, H, B), blk, 0, stream>>>(q_cat, kv_cat, v_b_w, v_out, S);
  // 8) out = v_out @ o_w^T            [4096,2048,2048]
  gemm_wmma_kernel<<<dim3(2048 / GBN, 4096 / GBM, 1), blk, 0, stream>>>(
      v_out, o_w, out, 4096, 2048, 2048, 2048, 2048, 2048, 1, 0, 1,
      0, 0, 0, 0, 0, 0);
}
